// MMD_loss_28827820491064
// MI455X (gfx1250) — matrix operand order repaired and checked by reading, not compiled, not run on hardware
//
#include <hip/hip_runtime.h>

// ---------------------------------------------------------------------------
// Fused MMD loss for MI455X (gfx1250, wave32, WMMA bf16 16x16x32).
//   1) sqnorm_kernel:   sq[i] = sum_k F[i,k]^2   (+ zero the 3 accumulators)
//   2) mmd_tile_kernel: 128x64 Gram tiles via v_wmma_f32_16x16x32_bf16 from
//                       LDS-staged bf16 tiles, fused d2 -> 12-sigma exp sum ->
//                       domain-masked partial sums -> 3 float atomics.
//   3) finalize_kernel: Ex - Ey scalar.
// ---------------------------------------------------------------------------

typedef __attribute__((ext_vector_type(16))) __bf16        v16bf;
typedef __attribute__((ext_vector_type(8)))  float         v8f;
typedef __attribute__((ext_vector_type(4)))  unsigned int  u32x4;
typedef __attribute__((ext_vector_type(2)))  unsigned int  u32x2;
typedef __attribute__((ext_vector_type(8)))  unsigned int  u32x8;
typedef __attribute__((ext_vector_type(4)))  float         f32x4;

__device__ __forceinline__ unsigned int f2bf(float f) {
    // round-to-nearest-even fp32 -> bf16
    unsigned int u = __float_as_uint(f);
    return (u + 0x7FFFu + ((u >> 16) & 1u)) >> 16;
}

__device__ __forceinline__ v16bf frag_from(u32x4 lo, u32x4 hi) {
    u32x8 w = {lo.x, lo.y, lo.z, lo.w, hi.x, hi.y, hi.z, hi.w};
    return __builtin_bit_cast(v16bf, w);
}

// --------------------------- kernel 1: row norms ---------------------------
__global__ void sqnorm_kernel(const float* __restrict__ F,
                              float* __restrict__ sq,
                              float* __restrict__ acc3,
                              int N, int D) {
    int i = blockIdx.x * blockDim.x + threadIdx.x;
    if (i == 0) { acc3[0] = 0.0f; acc3[1] = 0.0f; acc3[2] = 0.0f; }
    if (i < N) {
        const float* row = F + (size_t)i * D;
        float s = 0.0f;
        for (int k = 0; k < D; k += 4) {
            f32x4 f = *(const f32x4*)(row + k);
            s += f.x * f.x + f.y * f.y + f.z * f.z + f.w * f.w;
        }
        sq[i] = s;
    }
}

// ----------------------- kernel 2: fused MMD tiles -------------------------
__global__ __launch_bounds__(256) void mmd_tile_kernel(
    const float* __restrict__ F, const int* __restrict__ dom,
    const float* __restrict__ sq, float* __restrict__ acc3,
    int N, int D) {
    extern __shared__ char smem[];
    const int SE = D + 8;  // padded bf16 row stride (16B aligned, bank-spread)
    unsigned short* sA   = (unsigned short*)smem;          // 128 x SE bf16
    unsigned short* sB   = sA + 128 * SE;                  //  64 x SE bf16
    float*          sSqA = (float*)(sB + 64 * SE);         // 128
    float*          sSqB = sSqA + 128;                     //  64
    int*            sDomA = (int*)(sSqB + 64);             // 128
    int*            sDomB = sDomA + 128;                   //  64
    float*          sRed  = (float*)(sDomB + 64);          //  24

    const int tid  = threadIdx.x;
    const int lane = tid & 31;
    const int wv   = tid >> 5;
    const int rowBase = blockIdx.y * 128;
    const int colBase = blockIdx.x * 64;

    // ---- stage tiles (fp32 -> bf16, coalesced float4 loads) ----
    const int D4 = D >> 2;
    for (int v = tid; v < 128 * D4; v += 256) {
        int r  = v / D4;
        int c4 = v - r * D4;
        f32x4 f = ((const f32x4*)(F + (size_t)(rowBase + r) * D))[c4];
        u32x2 pk = {f2bf(f.x) | (f2bf(f.y) << 16), f2bf(f.z) | (f2bf(f.w) << 16)};
        *(u32x2*)(sA + r * SE + c4 * 4) = pk;
    }
    for (int v = tid; v < 64 * D4; v += 256) {
        int r  = v / D4;
        int c4 = v - r * D4;
        f32x4 f = ((const f32x4*)(F + (size_t)(colBase + r) * D))[c4];
        u32x2 pk = {f2bf(f.x) | (f2bf(f.y) << 16), f2bf(f.z) | (f2bf(f.w) << 16)};
        *(u32x2*)(sB + r * SE + c4 * 4) = pk;
    }
    if (tid < 128) { sSqA[tid] = sq[rowBase + tid]; sDomA[tid] = dom[rowBase + tid]; }
    if (tid < 64)  { sSqB[tid] = sq[colBase + tid]; sDomB[tid] = dom[colBase + tid]; }
    __syncthreads();

    // ---- WMMA: wave wv computes rows [16wv,16wv+16) x 64 cols ----
    v8f acc[4];
    acc[0] = (v8f)(0.0f); acc[1] = (v8f)(0.0f);
    acc[2] = (v8f)(0.0f); acc[3] = (v8f)(0.0f);

    const int aRow = 16 * wv + (lane & 15);
    const int aKb  = (lane >> 4) * 8;    // A: lanes>=16 hold K+8 group
    const int bKb  = (lane >> 4) * 8;    // B: lanes>=16 hold K+8 group

    for (int kk = 0; kk < D; kk += 32) {
        u32x4 alo = *(const u32x4*)(sA + aRow * SE + kk + aKb);
        u32x4 ahi = *(const u32x4*)(sA + aRow * SE + kk + aKb + 16);
        v16bf aF = frag_from(alo, ahi);
#pragma unroll
        for (int n = 0; n < 4; ++n) {
            int bRow = 16 * n + (lane & 15);
            u32x4 blo = *(const u32x4*)(sB + bRow * SE + kk + bKb);
            u32x4 bhi = *(const u32x4*)(sB + bRow * SE + kk + bKb + 16);
            v16bf bF = frag_from(blo, bhi);
            acc[n] = __builtin_amdgcn_wmma_f32_16x16x32_bf16(
                false, aF, false, bF, (short)0, acc[n], false, false);
        }
    }

    // ---- fused epilogue: d2 -> 12-sigma exp sum -> masked accumulate ----
    const float msig[12] = {-0.00005f, -0.0005f, -0.005f, -0.05f, -0.5f, -2.5f,
                            -5.0f, -7.5f, -10.0f, -12.5f, -15.0f, -50.0f};
    float pos = 0.0f, neg = 0.0f, cnt = 0.0f;
    const int halfOff = (lane >> 4) * 8;   // f32 C/D layout: M = r + 8*(lane>=16)
    const int jlBase  = lane & 15;         //                 N = lane & 15
#pragma unroll
    for (int n = 0; n < 4; ++n) {
        const int   jl  = 16 * n + jlBase;
        const float sqj = sSqB[jl];
        const int   dj  = sDomB[jl];
#pragma unroll
        for (int r = 0; r < 8; ++r) {
            const int il = 16 * wv + halfOff + r;
            float g  = acc[n][r];
            float d2 = fmaxf(sSqA[il] + sqj - 2.0f * g, 0.0f);
            float ks = 0.0f;
#pragma unroll
            for (int s = 0; s < 12; ++s) ks += __expf(msig[s] * d2);
            if (sDomA[il] == dj) { pos += ks; cnt += 1.0f; }
            else                 { neg += ks; }
        }
    }

    // ---- wave32 -> block -> global reduction ----
#pragma unroll
    for (int off = 16; off > 0; off >>= 1) {
        pos += __shfl_xor(pos, off);
        neg += __shfl_xor(neg, off);
        cnt += __shfl_xor(cnt, off);
    }
    if (lane == 0) { sRed[wv] = pos; sRed[8 + wv] = neg; sRed[16 + wv] = cnt; }
    __syncthreads();
    if (tid == 0) {
        float p = 0.0f, ng = 0.0f, c = 0.0f;
        for (int w = 0; w < 8; ++w) { p += sRed[w]; ng += sRed[8 + w]; c += sRed[16 + w]; }
        atomicAdd(&acc3[0], p);
        atomicAdd(&acc3[1], ng);
        atomicAdd(&acc3[2], c);
    }
}

// --------------------------- kernel 3: finalize ----------------------------
__global__ void finalize_kernel(const float* __restrict__ acc3,
                                float* __restrict__ out, float Ntot) {
    float pos = acc3[0], neg = acc3[1], cnt = acc3[2];
    out[0] = pos / cnt - neg / (Ntot - cnt);
}

// ------------------------------- launcher ----------------------------------
extern "C" void kernel_launch(void* const* d_in, const int* in_sizes, int n_in,
                              void* d_out, int out_size, void* d_ws, size_t ws_size,
                              hipStream_t stream) {
    const float* F   = (const float*)d_in[0];
    const int*   dom = (const int*)d_in[1];
    const int N = in_sizes[1];            // 4096
    const int D = in_sizes[0] / N;        // 256

    float* sq   = (float*)d_ws;           // N floats
    float* acc3 = sq + N;                 // pos, neg, cnt

    sqnorm_kernel<<<(N + 255) / 256, 256, 0, stream>>>(F, sq, acc3, N, D);

    const size_t SE  = (size_t)D + 8;
    const size_t lds = 192 * SE * 2       // bf16 A (128 rows) + B (64 rows)
                     + 192 * 4            // sq staging
                     + 192 * 4            // domain staging
                     + 24 * 4;            // reduction scratch
    dim3 grid(N / 64, N / 128);
    mmd_tile_kernel<<<grid, 256, lds, stream>>>(F, dom, sq, acc3, N, D);

    finalize_kernel<<<1, 1, 0, stream>>>(acc3, (float*)d_out, (float)N * (float)N);
}
